// GPTSelfAttention_33251636806132
// MI455X (gfx1250) — compile-verified, matrix operand order checked by reading, not run on hardware
//
#include <hip/hip_runtime.h>
#include <hip/hip_bf16.h>

#define HIDDEN 1024
#define NHEAD  16
#define HDIM   64
#define BATCH  2
#define SEQ    2048
#define MTOT   (BATCH*SEQ)   // 4096

typedef __bf16 bf16;
typedef __attribute__((ext_vector_type(16))) __bf16 v16bf;
typedef __attribute__((ext_vector_type(8)))  __bf16 bf16x8;
typedef __attribute__((ext_vector_type(4)))  __bf16 bf16x4;
typedef __attribute__((ext_vector_type(8)))  float  v8f;
typedef __attribute__((ext_vector_type(4)))  unsigned int v4u;
typedef __attribute__((ext_vector_type(8)))  int v8i;
typedef __attribute__((ext_vector_type(4)))  int v4i;

#if defined(__has_builtin)
#if __has_builtin(__builtin_amdgcn_tensor_load_to_lds) && __has_builtin(__builtin_amdgcn_s_wait_tensorcnt)
#define USE_TDM 1
#endif
#endif
#ifndef USE_TDM
#define USE_TDM 0
#endif

// LDS tile row pitches (elements): padded so 16-lane fragment reads hit 16
// distinct bank-quads (row pitch 144B / 80B -> start banks stride 36 / 20,
// gcd with 64 banks = 4 -> conflict-free b128 reads, 16B aligned rows).
#define KPITCH 72
#define VPITCH 40

union FragAB { v16bf v; bf16x8 h[2]; };
union AccU   { v8f v; float f[8]; };

__device__ __forceinline__ v16bf frag_load(const bf16* p) {
    FragAB f;
    f.h[0] = *(const bf16x8*)(p);
    f.h[1] = *(const bf16x8*)(p + 16);
    return f.v;
}

__device__ __forceinline__ v8f wmma_bf16(v16bf a, v16bf b, v8f c) {
    return __builtin_amdgcn_wmma_f32_16x16x32_bf16(false, a, false, b,
                                                   (short)0, c, false, false);
}

#if USE_TDM
// Build a 2D TDM descriptor (D#) and issue tensor_load_to_lds (6-arg form:
// g0, g1, g2, g3, extra group, cpol — higher groups zero => dims 2..4 unused).
// Group0: count=1 | lds_addr | global_addr | type=2.  Group1 per ISA 8.4.
__device__ __forceinline__ void tdm_load_2d(const bf16* gsrc, unsigned lds_addr,
                                            unsigned flags,
                                            unsigned tdim0, unsigned tdim1,
                                            unsigned tile0, unsigned tile1,
                                            unsigned stride0) {
    unsigned long long ga = (unsigned long long)gsrc;
    v4u g0;
    g0[0] = 1u;                                   // 1 valid descriptor
    g0[1] = lds_addr;                             // LDS byte address
    g0[2] = (unsigned)ga;                         // global_addr[31:0]
    g0[3] = (unsigned)(ga >> 32) | (2u << 30);    // global_addr[56:32] | type=2
    v8i g1;
    g1[0] = (int)flags;                           // data_size=2B, pad cfg
    g1[1] = (int)(tdim0 << 16);                   // tensor_dim0[15:0]
    g1[2] = (int)((tdim0 >> 16) | (tdim1 << 16)); // dim0[31:16] | dim1[15:0]
    g1[3] = (int)((tdim1 >> 16) | (tile0 << 16)); // dim1[31:16] | tile_dim0
    g1[4] = (int)(tile1 & 0xFFFFu);               // tile_dim1 (tile_dim2=0)
    g1[5] = (int)stride0;                         // tensor_dim0_stride[31:0]
    g1[6] = 0;                                    // stride0 hi | dim1_stride lo
    g1[7] = 0;
    v4i gz4 = {0, 0, 0, 0};
    v8i gz8 = {0, 0, 0, 0, 0, 0, 0, 0};
    __builtin_amdgcn_tensor_load_to_lds(g0, g1, gz4, gz4, gz8, 0);
}
// flags: ws_mask=0, data_size(2B)=1<<16, pad_enable=1<<20,
//        pad_interval(code c -> 2^(c+1) dwords), pad_amount(code a -> a+1 dwords)
#define KFLAGS (0x00010000u | 0x00100000u | (4u << 22) | (3u << 25)) // 32dw rows, +4dw pad
#define VFLAGS (0x00010000u | 0x00100000u | (3u << 22) | (3u << 25)) // 16dw rows, +4dw pad
#endif

// ---------------------------------------------------------------- prep: x -> bf16
__global__ __launch_bounds__(256) void conv_x_kernel(const float* __restrict__ x,
                                                     bf16* __restrict__ xb, int n) {
    int i = (blockIdx.x * blockDim.x + threadIdx.x) * 4;
    if (i + 3 < n) {
        float4 v = *(const float4*)(x + i);
        bf16x4 o;
        o.x = (bf16)v.x; o.y = (bf16)v.y; o.z = (bf16)v.z; o.w = (bf16)v.w;
        *(bf16x4*)(xb + i) = o;
    }
}

// -------------------------------------------- prep: W[k][n] fp32 -> Wt[n][k] bf16
__global__ __launch_bounds__(256) void wt_kernel(const float* __restrict__ Wq,
                                                 const float* __restrict__ Wk,
                                                 const float* __restrict__ Wv,
                                                 const float* __restrict__ Wo,
                                                 bf16* __restrict__ WqT,
                                                 bf16* __restrict__ WkT,
                                                 bf16* __restrict__ WvT,
                                                 bf16* __restrict__ WoT) {
    const float* W; bf16* T;
    switch (blockIdx.z) {
        case 0:  W = Wq; T = WqT; break;
        case 1:  W = Wk; T = WkT; break;
        case 2:  W = Wv; T = WvT; break;
        default: W = Wo; T = WoT; break;
    }
    __shared__ float tile[32][33];
    const int tx = threadIdx.x, ty = threadIdx.y;          // block (32,8)
    const int n0 = blockIdx.x * 32, k0 = blockIdx.y * 32;
    #pragma unroll
    for (int i = 0; i < 4; i++)
        tile[ty + 8 * i][tx] = W[(size_t)(k0 + ty + 8 * i) * HIDDEN + n0 + tx];
    __syncthreads();
    #pragma unroll
    for (int i = 0; i < 4; i++) {
        int n = n0 + ty + 8 * i;
        T[(size_t)n * HIDDEN + k0 + tx] = (bf16)tile[tx][ty + 8 * i];
    }
}

// ------------------------------------------------------- QKV projection GEMM
__global__ __launch_bounds__(256) void qkv_kernel(const bf16* __restrict__ X,
        const bf16* __restrict__ WqT, const bf16* __restrict__ WkT,
        const bf16* __restrict__ WvT,
        const float* __restrict__ bq, const float* __restrict__ bk,
        const float* __restrict__ bv,
        bf16* __restrict__ Qo, bf16* __restrict__ Ko, bf16* __restrict__ Vt) {
    const int mode = blockIdx.z;
    const bf16*  Bt   = (mode == 0) ? WqT : (mode == 1) ? WkT : WvT;
    const float* bias = (mode == 0) ? bq  : (mode == 1) ? bk  : bv;

    const int lane = threadIdx.x & 31;
    const int w    = threadIdx.x >> 5;
    const int lo   = lane & 15, hi = lane >> 4;
    const int m0   = blockIdx.y * 128 + (w & 1) * 64;
    const int n0   = blockIdx.x * 256 + (w >> 1) * 64;

    const v8f vz = {0.f, 0.f, 0.f, 0.f, 0.f, 0.f, 0.f, 0.f};
    v8f acc[4][4];
    #pragma unroll
    for (int i = 0; i < 4; i++)
        #pragma unroll
        for (int j = 0; j < 4; j++) acc[i][j] = vz;

    const bf16* Ab = X  + (size_t)(m0 + lo) * HIDDEN + hi * 8;
    const bf16* Bb = Bt + (size_t)(n0 + lo) * HIDDEN + hi * 8;

    for (int k0 = 0; k0 < HIDDEN; k0 += 32) {
        v16bf a[4], b[4];
        #pragma unroll
        for (int i = 0; i < 4; i++) a[i] = frag_load(Ab + (size_t)i * 16 * HIDDEN + k0);
        #pragma unroll
        for (int j = 0; j < 4; j++) b[j] = frag_load(Bb + (size_t)j * 16 * HIDDEN + k0);
        #pragma unroll
        for (int i = 0; i < 4; i++)
            #pragma unroll
            for (int j = 0; j < 4; j++)
                acc[i][j] = wmma_bf16(a[i], b[j], acc[i][j]);
    }

    #pragma unroll
    for (int i = 0; i < 4; i++) {
        #pragma unroll
        for (int j = 0; j < 4; j++) {
            const int n = n0 + j * 16 + lo;
            const float bval = bias[n];
            const int h = n >> 6, d = n & 63;
            AccU u; u.v = acc[i][j];
            #pragma unroll
            for (int r = 0; r < 8; r++) {
                const int m  = m0 + i * 16 + r + 8 * hi;
                const int bi = m >> 11, s = m & 2047;
                const float val = u.f[r] + bval;
                if (mode == 0)
                    Qo[((size_t)(bi * NHEAD + h) * SEQ + s) * HDIM + d] = (bf16)(val * 0.125f);
                else if (mode == 1)
                    Ko[((size_t)(bi * NHEAD + h) * SEQ + s) * HDIM + d] = (bf16)val;
                else
                    Vt[((size_t)(bi * NHEAD + h) * HDIM + d) * SEQ + s] = (bf16)val;
            }
        }
    }
}

// ------------------------------------------------------- fused causal flash attention
// Block = 256 q-rows of one head (8 waves x 32 rows). K/V tiles are staged into
// LDS once per block per 32-key step via the Tensor Data Mover (double-buffered,
// overlapped with compute); QK^T and P*V run on WMMA with online softmax.
__global__ __launch_bounds__(256) void flash_kernel(const bf16* __restrict__ Q,
                                                    const bf16* __restrict__ K,
                                                    const bf16* __restrict__ Vt,
                                                    bf16* __restrict__ AO) {
    __shared__ bf16 ldsK[2][32 * KPITCH];   // 32 keys x 64 d   (padded rows)
    __shared__ bf16 ldsV[2][64 * VPITCH];   // 64 d    x 32 keys (padded rows)
    __shared__ bf16 plds[8][32 * 32];       // per-wave P tile

    const int lane = threadIdx.x & 31;
    const int w    = threadIdx.x >> 5;
    const int lo   = lane & 15, hi = lane >> 4;
    const int h    = blockIdx.y, b = blockIdx.z;
    const int bh   = b * NHEAD + h;
    const int qb   = blockIdx.x * 256 + w * 32;
    const int qbLast = blockIdx.x * 256 + 224;

    const bf16* Qh = Q  + (size_t)bh * SEQ * HDIM;
    const bf16* Kh = K  + (size_t)bh * SEQ * HDIM;
    const bf16* Vh = Vt + (size_t)bh * HDIM * SEQ;

    // resident Q fragments: 2 M-tiles x 2 k-chunks (d = 0..31, 32..63)
    v16bf aq[2][2];
    #pragma unroll
    for (int mi = 0; mi < 2; mi++)
        #pragma unroll
        for (int kc = 0; kc < 2; kc++)
            aq[mi][kc] = frag_load(Qh + (size_t)(qb + mi * 16 + lo) * HDIM + kc * 32 + hi * 8);

    const v8f vz = {0.f, 0.f, 0.f, 0.f, 0.f, 0.f, 0.f, 0.f};
    v8f o[2][4];
    float mstat[2][8], lsum[2][8];
    #pragma unroll
    for (int mi = 0; mi < 2; mi++) {
        #pragma unroll
        for (int nj = 0; nj < 4; nj++) o[mi][nj] = vz;
        #pragma unroll
        for (int r = 0; r < 8; r++) { mstat[mi][r] = -3.0e38f; lsum[mi][r] = 0.f; }
    }

#if USE_TDM
    if (w == 0) {   // prologue: prefetch first K/V tile into buffer 0
        tdm_load_2d(Kh, (unsigned)(size_t)&ldsK[0][0], KFLAGS,
                    HDIM, SEQ, HDIM, 32, HDIM);
        tdm_load_2d(Vh, (unsigned)(size_t)&ldsV[0][0], VFLAGS,
                    SEQ, HDIM, 32, HDIM, SEQ);
    }
#endif

    for (int kb = 0; kb <= qbLast; kb += 32) {
        const int p = (kb >> 5) & 1;

#if USE_TDM
        if (w == 0) __builtin_amdgcn_s_wait_tensorcnt(0);
        __syncthreads();                       // buffer p is ready for everyone
        if (w == 0 && kb + 32 <= qbLast) {     // prefetch next tile into p^1
            const int nkb = kb + 32;
            tdm_load_2d(Kh + (size_t)nkb * HDIM, (unsigned)(size_t)&ldsK[p ^ 1][0],
                        KFLAGS, HDIM, SEQ, HDIM, 32, HDIM);
            tdm_load_2d(Vh + nkb, (unsigned)(size_t)&ldsV[p ^ 1][0],
                        VFLAGS, SEQ, HDIM, 32, HDIM, SEQ);
        }
#else
        {   // cooperative fallback: 256 threads load 8KB K/V tile
            const int t = threadIdx.x;
            {   // K: 32 rows x 8 chunks of 8 elems
                const int r = t >> 3, c = t & 7;
                *(bf16x8*)&ldsK[p][r * KPITCH + c * 8] =
                    *(const bf16x8*)(Kh + (size_t)(kb + r) * HDIM + c * 8);
            }
            {   // V: 64 rows x 4 chunks of 8 elems
                const int r = t >> 2, c = t & 3;
                *(bf16x8*)&ldsV[p][r * VPITCH + c * 8] =
                    *(const bf16x8*)(Vh + (size_t)r * SEQ + kb + c * 8);
            }
            __syncthreads();
        }
#endif

        if (kb <= qb) {   // wave-uniform predicate: EXEC stays all-ones for WMMA
            const bool diag = (kb == qb);

            // ---- S = Q * K^T  (32q x 32k) from LDS K tile
            v8f sa[2][2] = {{vz, vz}, {vz, vz}};
            #pragma unroll
            for (int kc = 0; kc < 2; kc++) {
                v16bf bk0 = frag_load(&ldsK[p][(lo)      * KPITCH + kc * 32 + hi * 8]);
                v16bf bk1 = frag_load(&ldsK[p][(16 + lo) * KPITCH + kc * 32 + hi * 8]);
                sa[0][0] = wmma_bf16(aq[0][kc], bk0, sa[0][0]);
                sa[0][1] = wmma_bf16(aq[0][kc], bk1, sa[0][1]);
                sa[1][0] = wmma_bf16(aq[1][kc], bk0, sa[1][0]);
                sa[1][1] = wmma_bf16(aq[1][kc], bk1, sa[1][1]);
            }

            // ---- causal mask + online softmax; write P tile (bf16) to wave LDS
            #pragma unroll
            for (int mi = 0; mi < 2; mi++) {
                AccU u0, u1, facu;
                u0.v = sa[mi][0]; u1.v = sa[mi][1];
                #pragma unroll
                for (int r = 0; r < 8; r++) {
                    const int qi = qb + mi * 16 + r + 8 * hi;
                    if (diag) {
                        if (kb + lo      > qi) u0.f[r] = -100000.f;
                        if (kb + 16 + lo > qi) u1.f[r] = -100000.f;
                    }
                    float rm = fmaxf(u0.f[r], u1.f[r]);
                    #pragma unroll
                    for (int off = 1; off < 16; off <<= 1)
                        rm = fmaxf(rm, __shfl_xor(rm, off, 32));
                    const float mnew = fmaxf(mstat[mi][r], rm);
                    const float fac  = __expf(mstat[mi][r] - mnew);
                    const float p0   = __expf(u0.f[r] - mnew);
                    const float p1   = __expf(u1.f[r] - mnew);
                    float ps = p0 + p1;
                    #pragma unroll
                    for (int off = 1; off < 16; off <<= 1)
                        ps += __shfl_xor(ps, off, 32);
                    lsum[mi][r]  = lsum[mi][r] * fac + ps;
                    mstat[mi][r] = mnew;
                    facu.f[r]    = fac;
                    const int row = mi * 16 + r + 8 * hi;
                    plds[w][row * 32 + lo]      = (bf16)p0;
                    plds[w][row * 32 + 16 + lo] = (bf16)p1;
                }
                #pragma unroll
                for (int nj = 0; nj < 4; nj++) o[mi][nj] = o[mi][nj] * facu.v;
            }

            // wave-private P tile; DS ops in-order per wave, fence HW + compiler
            asm volatile("s_wait_dscnt 0" ::: "memory");

            // ---- O += P * V  (K = 32 keys) from LDS V tile
            v16bf bvf[4];
            #pragma unroll
            for (int nj = 0; nj < 4; nj++)
                bvf[nj] = frag_load(&ldsV[p][(nj * 16 + lo) * VPITCH + hi * 8]);
            #pragma unroll
            for (int mi = 0; mi < 2; mi++) {
                const bf16* pp = &plds[w][(mi * 16 + lo) * 32 + hi * 8];
                FragAB pf;
                pf.h[0] = *(const bf16x8*)(pp);
                pf.h[1] = *(const bf16x8*)(pp + 16);
                #pragma unroll
                for (int nj = 0; nj < 4; nj++)
                    o[mi][nj] = wmma_bf16(pf.v, bvf[nj], o[mi][nj]);
            }
        }
        __syncthreads();   // everyone done reading buffer p before it is reused
    }

    // ---- normalize and store attention output as [b][s][h*64+d] bf16
    #pragma unroll
    for (int mi = 0; mi < 2; mi++) {
        AccU lr;
        #pragma unroll
        for (int r = 0; r < 8; r++) lr.f[r] = 1.f / lsum[mi][r];
        #pragma unroll
        for (int nj = 0; nj < 4; nj++) {
            AccU u; u.v = o[mi][nj] * lr.v;
            const int d = nj * 16 + lo;
            #pragma unroll
            for (int r = 0; r < 8; r++) {
                const int s = qb + mi * 16 + r + 8 * hi;
                AO[((size_t)(b * SEQ + s) * HIDDEN) + h * HDIM + d] = (bf16)u.f[r];
            }
        }
    }
}

// ------------------------------------------------------- output projection GEMM
__global__ __launch_bounds__(256) void oproj_kernel(const bf16* __restrict__ AO,
                                                    const bf16* __restrict__ WoT,
                                                    const float* __restrict__ bo,
                                                    float* __restrict__ out) {
    const int lane = threadIdx.x & 31;
    const int w    = threadIdx.x >> 5;
    const int lo   = lane & 15, hi = lane >> 4;
    const int m0   = blockIdx.y * 128 + (w & 1) * 64;
    const int n0   = blockIdx.x * 256 + (w >> 1) * 64;

    const v8f vz = {0.f, 0.f, 0.f, 0.f, 0.f, 0.f, 0.f, 0.f};
    v8f acc[4][4];
    #pragma unroll
    for (int i = 0; i < 4; i++)
        #pragma unroll
        for (int j = 0; j < 4; j++) acc[i][j] = vz;

    const bf16* Ab = AO  + (size_t)(m0 + lo) * HIDDEN + hi * 8;
    const bf16* Bb = WoT + (size_t)(n0 + lo) * HIDDEN + hi * 8;

    for (int k0 = 0; k0 < HIDDEN; k0 += 32) {
        v16bf a[4], b[4];
        #pragma unroll
        for (int i = 0; i < 4; i++) a[i] = frag_load(Ab + (size_t)i * 16 * HIDDEN + k0);
        #pragma unroll
        for (int j = 0; j < 4; j++) b[j] = frag_load(Bb + (size_t)j * 16 * HIDDEN + k0);
        #pragma unroll
        for (int i = 0; i < 4; i++)
            #pragma unroll
            for (int j = 0; j < 4; j++)
                acc[i][j] = wmma_bf16(a[i], b[j], acc[i][j]);
    }

    #pragma unroll
    for (int i = 0; i < 4; i++) {
        #pragma unroll
        for (int j = 0; j < 4; j++) {
            const int n = n0 + j * 16 + lo;
            const float bval = bo[n];
            AccU u; u.v = acc[i][j];
            #pragma unroll
            for (int r = 0; r < 8; r++) {
                const int m = m0 + i * 16 + r + 8 * hi;
                out[(size_t)m * HIDDEN + n] = u.f[r] + bval;
            }
        }
    }
}

// ----------------------------------------------------------------------- launch
extern "C" void kernel_launch(void* const* d_in, const int* in_sizes, int n_in,
                              void* d_out, int out_size, void* d_ws, size_t ws_size,
                              hipStream_t stream) {
    (void)in_sizes; (void)n_in; (void)out_size; (void)ws_size;

    const float* x  = (const float*)d_in[0];
    const float* Wq = (const float*)d_in[1];
    const float* bq = (const float*)d_in[2];
    const float* Wk = (const float*)d_in[3];
    const float* bk = (const float*)d_in[4];
    const float* Wv = (const float*)d_in[5];
    const float* bv = (const float*)d_in[6];
    const float* Wo = (const float*)d_in[7];
    const float* bo = (const float*)d_in[8];

    const size_t SZ = (size_t)MTOT * HIDDEN;       // 4194304 elements
    const size_t WZ = (size_t)HIDDEN * HIDDEN;     // 1048576 elements
    bf16* Xb  = (bf16*)d_ws;
    bf16* WqT = Xb  + SZ;
    bf16* WkT = WqT + WZ;
    bf16* WvT = WkT + WZ;
    bf16* WoT = WvT + WZ;
    bf16* Qb  = WoT + WZ;
    bf16* Kb  = Qb  + SZ;
    bf16* Vtb = Kb  + SZ;
    bf16* AO  = Vtb + SZ;

    conv_x_kernel<<<4096, 256, 0, stream>>>(x, Xb, (int)SZ);
    wt_kernel<<<dim3(32, 32, 4), dim3(32, 8), 0, stream>>>(Wq, Wk, Wv, Wo,
                                                           WqT, WkT, WvT, WoT);
    qkv_kernel<<<dim3(4, 32, 3), 256, 0, stream>>>(Xb, WqT, WkT, WvT,
                                                   bq, bk, bv, Qb, Kb, Vtb);
    flash_kernel<<<dim3(8, NHEAD, BATCH), 256, 0, stream>>>(Qb, Kb, Vtb, AO);
    oproj_kernel<<<dim3(4, 32), 256, 0, stream>>>(AO, WoT, bo, (float*)d_out);
}
